// DualAttentionModule_50319836839988
// MI455X (gfx1250) — matrix operand order repaired and checked by reading, not compiled, not run on hardware
//
#include <hip/hip_runtime.h>

// ---------------------------------------------------------------------------
// DualAttention (channel attention) for MI455X / gfx1250, wave32 + WMMA bf16.
//   B=16, H=W=64, C=512, N=H*W=4096, conv K=3 (SAME), im2col K-dim = 4608.
// Matrix math via v_wmma_f32_16x16x32_bf16 (f32 accumulate); kernel 4 stages
// its LDS tiles with the Tensor Data Mover (tensor_load_to_lds + TENSORcnt).
// ---------------------------------------------------------------------------

typedef __bf16 bf16_t;
typedef __attribute__((ext_vector_type(16))) __bf16 v16bf;
typedef __attribute__((ext_vector_type(8)))  __bf16 v8bf;
typedef __attribute__((ext_vector_type(8)))  float  v8f;
typedef __attribute__((ext_vector_type(4)))  unsigned int u32x4;
typedef __attribute__((ext_vector_type(4)))  int i32x4;
typedef __attribute__((ext_vector_type(8)))  int i32x8;

#define C_DIM 512
#define HW_DIM 4096          // 64*64
#define BATCH 16
#define PER_BATCH (C_DIM * HW_DIM)   // 2097152

// round-to-nearest-even f32 -> bf16
__device__ __forceinline__ bf16_t f2bf(float f) {
  unsigned u = __builtin_bit_cast(unsigned, f);
  unsigned r = u + 0x7FFFu + ((u >> 16) & 1u);
  unsigned short h = (unsigned short)(r >> 16);
  return __builtin_bit_cast(bf16_t, h);
}

__device__ __forceinline__ v16bf cat8(v8bf lo, v8bf hi) {
  return __builtin_shufflevector(lo, hi, 0, 1, 2, 3, 4, 5, 6, 7,
                                 8, 9, 10, 11, 12, 13, 14, 15);
}

// A fragment (16x32 bf16, MxK): lane<16 holds K{0..7,16..23}, lane>=16 K{8..15,24..31}
__device__ __forceinline__ v16bf load_a_frag(const bf16_t* row, int lane) {
  const int kb = (lane & 16) ? 8 : 0;
  v8bf lo = *(const v8bf*)(row + kb);
  v8bf hi = *(const v8bf*)(row + 16 + kb);
  return cat8(lo, hi);
}

// B fragment (32x16 bf16, KxN), staged K-contiguous per output column:
// lane<16 holds K{0..7,16..23}, lane>=16 holds K{8..15,24..31} of column (lane&15)
__device__ __forceinline__ v16bf load_b_frag(const bf16_t* row, int lane) {
  const int kb = (lane & 16) ? 8 : 0;
  v8bf lo = *(const v8bf*)(row + kb);
  v8bf hi = *(const v8bf*)(row + 16 + kb);
  return cat8(lo, hi);
}

// One K=32 step: wave computes a 32(M) x 64(N) patch = 2x4 WMMA tiles.
// Both LDS tiles are laid out [out-dim][32 K] so fragments are 16B ds loads.
__device__ __forceinline__ void mma_tile(const bf16_t (*lA)[32], const bf16_t (*lB)[32],
                                         int mBase, int nBase, int lane, v8f acc[2][4]) {
  v16bf af[2], bfr[4];
#pragma unroll
  for (int tm = 0; tm < 2; ++tm)
    af[tm] = load_a_frag(lA[mBase + 16 * tm + (lane & 15)], lane);
#pragma unroll
  for (int tn = 0; tn < 4; ++tn)
    bfr[tn] = load_b_frag(lB[nBase + 16 * tn + (lane & 15)], lane);
#pragma unroll
  for (int tm = 0; tm < 2; ++tm)
#pragma unroll
    for (int tn = 0; tn < 4; ++tn)
      acc[tm][tn] = __builtin_amdgcn_wmma_f32_16x16x32_bf16(
          false, af[tm], false, bfr[tn], (short)0, acc[tm][tn], false, false);
}

// ---------------------------------------------------------------------------
// TDM: issue one 2-D tensor_load_to_lds (tile = tile_x elems  x  tile_y rows,
// element = 2 bytes, row stride = strideElems). D# built per CDNA5 ISA 8.3/8.4.
// Must be executed by a single wave; descriptor values are uniform.
// ---------------------------------------------------------------------------
__device__ __forceinline__ void tdm_load_2d(unsigned ldsOff, const void* gptr,
                                            unsigned tileX, unsigned tileY,
                                            unsigned strideElems) {
  const unsigned long long ga = (unsigned long long)gptr;
  u32x4 g0;
  g0[0] = 1u;                                   // count=1, user descriptor
  g0[1] = ldsOff;                               // lds_addr (bytes)
  g0[2] = (unsigned)(ga & 0xFFFFFFFFu);         // global_addr[31:0]
  g0[3] = (unsigned)((ga >> 32) & 0x01FFFFFFu)  // global_addr[56:32]
          | (2u << 30);                         // type = 2 ("image")
  i32x8 g1;
  g1[0] = (int)(1u << 16);                      // data_size=1 (2B), no multicast
  g1[1] = (int)((strideElems & 0xFFFFu) << 16); // tensor_dim0[15:0]
  g1[2] = (int)((strideElems >> 16) & 0xFFFFu)  // tensor_dim0[31:16]
          | (int)(0xFFFFu << 16);               // tensor_dim1[15:0] (huge, no OOB)
  g1[3] = (int)((tileX & 0xFFFFu) << 16);       // tensor_dim1[31:16]=0 | tile_dim0
  g1[4] = (int)(tileY & 0xFFFFu);               // tile_dim1 | tile_dim2=0
  g1[5] = (int)strideElems;                     // tensor_dim0_stride[31:0]
  g1[6] = 0;                                    // stride hi | dim1_stride lo
  g1[7] = 0;
  const i32x4 z4 = {};
#if __clang_major__ >= 23
  const i32x8 z8 = {};
  __builtin_amdgcn_tensor_load_to_lds(g0, g1, z4, z4, z8, 0);
#else
  __builtin_amdgcn_tensor_load_to_lds(g0, g1, z4, z4, 0);
#endif
}

// ---------------------------------------------------------------------------
// Kernel 1: fused 3x3 conv + BN(inference) + ReLU, implicit GEMM.
//   M = B*H*W = 65536 (grid.x = 512 tiles of 128)
//   N = Cout  = 512   (grid.y = 4 tiles of 128)
//   K = 9*512 = 4608  (144 steps of 32; each step within one filter tap)
// Output bf16 [B*HW, C] (== NHWC flatten).
// ---------------------------------------------------------------------------
__global__ __launch_bounds__(256)
void conv_bn_relu_kernel(const float* __restrict__ x, const float* __restrict__ W,
                         const float* __restrict__ bias, const float* __restrict__ gamma,
                         const float* __restrict__ beta, const float* __restrict__ mean,
                         const float* __restrict__ var, bf16_t* __restrict__ out) {
  __shared__ bf16_t lA[128][32];   // x im2col tile  [m][k]
  __shared__ bf16_t lB[128][32];   // weight tile    [cout][cin-chunk]
  const int tid = threadIdx.x;
  const int lane = tid & 31, wave = tid >> 5;
  const int m0 = blockIdx.x * 128;
  const int n0 = blockIdx.y * 128;
  const int waveM = wave & 3, waveN = wave >> 2;   // 4x2 wave grid -> 32x64 each

  v8f acc[2][4] = {};

  // A-loader mapping: 128 rows x 2 channel-halves of 16
  const int mloc = tid & 127, cb = (tid >> 7) << 4;
  const int m = m0 + mloc;
  const int bb = m >> 12, hh = (m >> 6) & 63, ww = m & 63;
  // B-loader mapping: 32 cin x 8 groups of 16 cout (transpose into LDS)
  const int kloc = tid & 31, nbase = (tid >> 5) << 4;

  for (int ks = 0; ks < 144; ++ks) {
    const int tap = ks >> 4;              // ky*3+kx
    const int c0 = (ks & 15) << 5;        // cin base within tap
    const int dy = tap / 3 - 1, dx = tap % 3 - 1;

    // ---- stage A: x with SAME-padding halo, f32 -> bf16 ----
    {
      const int hin = hh + dy, win = ww + dx;
      v8bf r0 = {}, r1 = {};
      if (((unsigned)hin < 64u) && ((unsigned)win < 64u)) {
        const float* src = x + (((size_t)bb * 64 + hin) * 64 + win) * 512 + c0 + cb;
        float4 f0 = *(const float4*)(src + 0);
        float4 f1 = *(const float4*)(src + 4);
        float4 f2 = *(const float4*)(src + 8);
        float4 f3 = *(const float4*)(src + 12);
        r0[0] = f2bf(f0.x); r0[1] = f2bf(f0.y); r0[2] = f2bf(f0.z); r0[3] = f2bf(f0.w);
        r0[4] = f2bf(f1.x); r0[5] = f2bf(f1.y); r0[6] = f2bf(f1.z); r0[7] = f2bf(f1.w);
        r1[0] = f2bf(f2.x); r1[1] = f2bf(f2.y); r1[2] = f2bf(f2.z); r1[3] = f2bf(f2.w);
        r1[4] = f2bf(f3.x); r1[5] = f2bf(f3.y); r1[6] = f2bf(f3.z); r1[7] = f2bf(f3.w);
      }
      *(v8bf*)&lA[mloc][cb] = r0;
      *(v8bf*)&lA[mloc][cb + 8] = r1;
    }
    // ---- stage B: weights HWIO, transpose cin<->cout into [cout][cin] ----
    {
      const float* wsrc = W + ((size_t)tap * 512 + (c0 + kloc)) * 512 + n0 + nbase;
      float4 g0 = *(const float4*)(wsrc + 0);
      float4 g1 = *(const float4*)(wsrc + 4);
      float4 g2 = *(const float4*)(wsrc + 8);
      float4 g3 = *(const float4*)(wsrc + 12);
      lB[nbase + 0][kloc] = f2bf(g0.x);  lB[nbase + 1][kloc] = f2bf(g0.y);
      lB[nbase + 2][kloc] = f2bf(g0.z);  lB[nbase + 3][kloc] = f2bf(g0.w);
      lB[nbase + 4][kloc] = f2bf(g1.x);  lB[nbase + 5][kloc] = f2bf(g1.y);
      lB[nbase + 6][kloc] = f2bf(g1.z);  lB[nbase + 7][kloc] = f2bf(g1.w);
      lB[nbase + 8][kloc] = f2bf(g2.x);  lB[nbase + 9][kloc] = f2bf(g2.y);
      lB[nbase + 10][kloc] = f2bf(g2.z); lB[nbase + 11][kloc] = f2bf(g2.w);
      lB[nbase + 12][kloc] = f2bf(g3.x); lB[nbase + 13][kloc] = f2bf(g3.y);
      lB[nbase + 14][kloc] = f2bf(g3.z); lB[nbase + 15][kloc] = f2bf(g3.w);
    }
    __syncthreads();
    mma_tile(lA, lB, waveM * 32, waveN * 64, lane, acc);
    __syncthreads();
  }

  // ---- epilogue: BN fold + ReLU, store bf16 ----
#pragma unroll
  for (int tn = 0; tn < 4; ++tn) {
    const int cch = n0 + waveN * 64 + 16 * tn + (lane & 15);
    const float sc = gamma[cch] * rsqrtf(var[cch] + 1e-3f);
    const float sh = (bias[cch] - mean[cch]) * sc + beta[cch];
#pragma unroll
    for (int tm = 0; tm < 2; ++tm) {
      const int rowb = m0 + waveM * 32 + 16 * tm + ((lane >> 4) << 3);
#pragma unroll
      for (int r = 0; r < 8; ++r) {
        float y = fmaxf(fmaf(acc[tm][tn][r], sc, sh), 0.f);
        out[(size_t)(rowb + r) * 512 + cch] = f2bf(y);
      }
    }
  }
}

// ---------------------------------------------------------------------------
// Kernel 2: energy[b,c,d] = sum_n q[b,n,c]*k[b,n,d]  (A^T * B, K = 4096)
// grid: (c-tiles=4, d-tiles=4, B=16)
// ---------------------------------------------------------------------------
__global__ __launch_bounds__(256)
void qk_energy_kernel(const bf16_t* __restrict__ q, const bf16_t* __restrict__ k,
                      float* __restrict__ energy) {
  __shared__ bf16_t lA[128][32];
  __shared__ bf16_t lB[128][32];
  const int tid = threadIdx.x, lane = tid & 31, wave = tid >> 5;
  const int c0 = blockIdx.x * 128, d0 = blockIdx.y * 128;
  const int b = blockIdx.z;
  const bf16_t* qb = q + (size_t)b * HW_DIM * 512;
  const bf16_t* kmb = k + (size_t)b * HW_DIM * 512;
  const int waveM = wave & 3, waveN = wave >> 2;
  v8f acc[2][4] = {};

  // loader mapping: 32 k-rows x 8 groups of 16 output-dim (transpose into LDS)
  const int kloc = tid & 31, obase = (tid >> 5) << 4;

  for (int ks = 0; ks < 128; ++ks) {
    const int nk0 = ks << 5;
    {
      const bf16_t* src = qb + (size_t)(nk0 + kloc) * 512 + c0 + obase;
      v8bf a0 = *(const v8bf*)(src);
      v8bf a1 = *(const v8bf*)(src + 8);
#pragma unroll
      for (int i = 0; i < 8; ++i) { lA[obase + i][kloc] = a0[i]; lA[obase + 8 + i][kloc] = a1[i]; }
    }
    {
      const bf16_t* src = kmb + (size_t)(nk0 + kloc) * 512 + d0 + obase;
      v8bf b0 = *(const v8bf*)(src);
      v8bf b1 = *(const v8bf*)(src + 8);
#pragma unroll
      for (int i = 0; i < 8; ++i) { lB[obase + i][kloc] = b0[i]; lB[obase + 8 + i][kloc] = b1[i]; }
    }
    if (ks + 1 < 128) {   // global_prefetch_b8 for the next K slab
      __builtin_prefetch(qb + (size_t)(nk0 + 32 + kloc) * 512 + c0 + obase, 0, 0);
      __builtin_prefetch(kmb + (size_t)(nk0 + 32 + kloc) * 512 + d0 + obase, 0, 0);
    }
    __syncthreads();
    mma_tile(lA, lB, waveM * 32, waveN * 64, lane, acc);
    __syncthreads();
  }

  float* eb = energy + (size_t)b * 512 * 512;
#pragma unroll
  for (int tn = 0; tn < 4; ++tn) {
    const int d = d0 + waveN * 64 + 16 * tn + (lane & 15);
#pragma unroll
    for (int tm = 0; tm < 2; ++tm) {
      const int crow = c0 + waveM * 32 + 16 * tm + ((lane >> 4) << 3);
#pragma unroll
      for (int r = 0; r < 8; ++r)
        eb[(size_t)(crow + r) * 512 + d] = acc[tm][tn][r];
    }
  }
}

// ---------------------------------------------------------------------------
// Kernel 3: row softmax over energy [8192 rows x 512], one wave32 per row.
// ---------------------------------------------------------------------------
__global__ __launch_bounds__(256)
void softmax_kernel(const float* __restrict__ energy, bf16_t* __restrict__ attn) {
  const int lane = threadIdx.x & 31;
  const int wave = threadIdx.x >> 5;
  const int row = blockIdx.x * 8 + wave;
  const float* e = energy + (size_t)row * 512;
  float vals[16];
  float mx = -3.4e38f;
#pragma unroll
  for (int i = 0; i < 16; ++i) { vals[i] = e[lane + 32 * i]; mx = fmaxf(mx, vals[i]); }
#pragma unroll
  for (int s = 16; s > 0; s >>= 1) mx = fmaxf(mx, __shfl_xor(mx, s, 32));
  float sum = 0.f;
#pragma unroll
  for (int i = 0; i < 16; ++i) { vals[i] = __expf(vals[i] - mx); sum += vals[i]; }
#pragma unroll
  for (int s = 16; s > 0; s >>= 1) sum += __shfl_xor(sum, s, 32);
  const float inv = 1.f / sum;
  bf16_t* a = attn + (size_t)row * 512;
#pragma unroll
  for (int i = 0; i < 16; ++i) a[lane + 32 * i] = f2bf(vals[i] * inv);
}

// ---------------------------------------------------------------------------
// Kernel 4: out[b,c,n] = sum_d attn[b,c,d]*v[b,n,d], fused aw*out + x.
// Both operands are K-contiguous 2-D strided tiles -> staged with the
// Tensor Data Mover (tensor_load_to_lds, one op per tile per K-step, issued
// by wave 0, synchronized with s_wait_tensorcnt + workgroup barrier).
// grid: (n-tiles=32, c-tiles=4, B=16)
// ---------------------------------------------------------------------------
struct Smem4 {
  bf16_t A[128][32];   // attn tile, LDS offset 0
  bf16_t B[128][32];   // v tile,    LDS offset 8192
};

__global__ __launch_bounds__(256)
void attn_v_residual_kernel(const bf16_t* __restrict__ attn, const bf16_t* __restrict__ v,
                            const float* __restrict__ x, const float* __restrict__ aw,
                            float* __restrict__ out) {
  __shared__ Smem4 sm;
  const int tid = threadIdx.x, lane = tid & 31, wave = tid >> 5;
  const int n0 = blockIdx.x * 128;
  const int c0 = blockIdx.y * 128;
  const int b = blockIdx.z;
  const bf16_t* ab = attn + (size_t)b * 512 * 512;
  const bf16_t* vb = v + (size_t)b * HW_DIM * 512;
  const int waveM = wave & 3, waveN = wave >> 2;
  v8f acc[2][4] = {};

  for (int ks = 0; ks < 16; ++ks) {
    const int k0 = ks << 5;
    __syncthreads();                 // prior WMMA reads done before TDM overwrite
    if (wave == 0) {
      // A tile: attn[c0 .. c0+127][k0 .. k0+31]  (row stride 512 elems)
      tdm_load_2d(0u, ab + (size_t)c0 * 512 + k0, 32u, 128u, 512u);
      // B tile: v[n0 .. n0+127][k0 .. k0+31]
      tdm_load_2d(8192u, vb + (size_t)n0 * 512 + k0, 32u, 128u, 512u);
      __builtin_amdgcn_s_wait_tensorcnt(0);
    }
    __syncthreads();                 // TDM data visible to all waves
    mma_tile(sm.A, sm.B, waveM * 32, waveN * 64, lane, acc);
  }

  const float a0 = aw[0];
  const float* xb = x + (size_t)b * PER_BATCH;
  float* ob = out + (size_t)b * PER_BATCH;
#pragma unroll
  for (int tn = 0; tn < 4; ++tn) {
    const int n = n0 + waveN * 64 + 16 * tn + (lane & 15);
#pragma unroll
    for (int tm = 0; tm < 2; ++tm) {
      const int crow = c0 + waveM * 32 + 16 * tm + ((lane >> 4) << 3);
#pragma unroll
      for (int r = 0; r < 8; ++r) {
        const size_t f = (size_t)(crow + r) * HW_DIM + n;  // raw reshape index
        ob[f] = fmaf(a0, acc[tm][tn][r], xb[f]);
      }
    }
  }
}

// ---------------------------------------------------------------------------
extern "C" void kernel_launch(void* const* d_in, const int* in_sizes, int n_in,
                              void* d_out, int out_size, void* d_ws, size_t ws_size,
                              hipStream_t stream) {
  const float* x  = (const float*)d_in[0];
  const float* Wq = (const float*)d_in[1];
  const float* bq = (const float*)d_in[2];
  const float* Wk = (const float*)d_in[3];
  const float* bk = (const float*)d_in[4];
  const float* Wv = (const float*)d_in[5];
  const float* bv = (const float*)d_in[6];
  const float* qg = (const float*)d_in[7];  const float* qbe = (const float*)d_in[8];
  const float* qm = (const float*)d_in[9];  const float* qv  = (const float*)d_in[10];
  const float* kg = (const float*)d_in[11]; const float* kbe = (const float*)d_in[12];
  const float* km = (const float*)d_in[13]; const float* kv  = (const float*)d_in[14];
  const float* vg = (const float*)d_in[15]; const float* vbe = (const float*)d_in[16];
  const float* vm = (const float*)d_in[17]; const float* vv  = (const float*)d_in[18];
  const float* aw = (const float*)d_in[19];
  float* out = (float*)d_out;

  // workspace layout (bytes): q 64MB | k 64MB | v 64MB | energy 16MB
  char* ws = (char*)d_ws;
  bf16_t* q    = (bf16_t*)(ws);
  bf16_t* kmat = (bf16_t*)(ws + (size_t)64 * 1024 * 1024);
  bf16_t* vmat = (bf16_t*)(ws + (size_t)128 * 1024 * 1024);
  float* energy = (float*)(ws + (size_t)192 * 1024 * 1024);
  bf16_t* attn = (bf16_t*)(ws);   // reuse q region (dead after energy GEMM)

  const dim3 blk(256);
  const dim3 gConv(512, 4, 1);
  conv_bn_relu_kernel<<<gConv, blk, 0, stream>>>(x, Wq, bq, qg, qbe, qm, qv, q);
  conv_bn_relu_kernel<<<gConv, blk, 0, stream>>>(x, Wk, bk, kg, kbe, km, kv, kmat);
  conv_bn_relu_kernel<<<gConv, blk, 0, stream>>>(x, Wv, bv, vg, vbe, vm, vv, vmat);
  qk_energy_kernel<<<dim3(4, 4, 16), blk, 0, stream>>>(q, kmat, energy);
  softmax_kernel<<<dim3(1024), blk, 0, stream>>>(energy, attn);
  attn_v_residual_kernel<<<dim3(32, 4, 16), blk, 0, stream>>>(attn, vmat, x, aw, out);
}
